// BoxHead_60997125537856
// MI455X (gfx1250) — compile-verified, matrix operand order checked by reading, not run on hardware
//
#include <hip/hip_runtime.h>

// ---------------------------------------------------------------------------
// Types
// ---------------------------------------------------------------------------
typedef __attribute__((ext_vector_type(16))) __bf16 v16bf;
typedef __attribute__((ext_vector_type(8)))  __bf16 v8bf;
typedef __attribute__((ext_vector_type(4)))  __bf16 v4bf;
typedef __attribute__((ext_vector_type(8)))  float  v8f;

typedef __attribute__((address_space(3))) const char lds_char;

__device__ __forceinline__ unsigned lds_off(const void* p) {
    // generic -> LDS addrspace cast, then ptrtoint: byte offset for async-to-LDS ops
    return (unsigned)(unsigned long long)(lds_char*)p;
}

// Build one v16bf WMMA fragment register set from two contiguous 16B chunks
// at p[0..7] and p[16..23] (matches the 16-bit A/B VGPR layout: lanes pick kb).
__device__ __forceinline__ v16bf load_frag(const __bf16* p) {
    v8bf lo = *(const v8bf*)(p);
    v8bf hi = *(const v8bf*)(p + 16);
    return __builtin_shufflevector(lo, hi, 0,1,2,3,4,5,6,7,8,9,10,11,12,13,14,15);
}

// Issue async global->LDS copy of one 128x32 bf16 tile (8 KB): 512 x 16B chunks,
// 2 per thread. LDS rows padded to 80 bytes (40 bf16) to avoid bank conflicts.
__device__ __forceinline__ void issue_tile(const void* base, int rowBytes,
                                           unsigned ldsBase, int kByteOff, int tid) {
#pragma unroll
    for (int t = 0; t < 2; ++t) {
        int c   = tid + t * 256;
        int row = c >> 2;
        int seg = c & 3;
        unsigned lds = ldsBase + (unsigned)(row * 80 + seg * 16);
        unsigned gof = (unsigned)(row * rowBytes + kByteOff + seg * 16);
        asm volatile("global_load_async_to_lds_b128 %0, %1, %2"
                     :: "v"(lds), "v"(gof), "s"(base) : "memory");
    }
}

// ---------------------------------------------------------------------------
// fp32 -> bf16 bulk convert (vectorized by 4)
// ---------------------------------------------------------------------------
__global__ __launch_bounds__(256) void cvt_bf16_4(const float* __restrict__ in,
                                                  unsigned short* __restrict__ out_,
                                                  long n4) {
    __bf16* out = (__bf16*)out_;
    long i = (long)blockIdx.x * 256 + threadIdx.x;
    if (i < n4) {
        float4 v = ((const float4*)in)[i];
        v4bf r;
        r.x = (__bf16)v.x; r.y = (__bf16)v.y; r.z = (__bf16)v.z; r.w = (__bf16)v.w;
        ((v4bf*)out)[i] = r;
    }
}

// ---------------------------------------------------------------------------
// Transpose + convert: in[R][C] fp32 -> out[C][R] bf16 (R,C multiples of 32)
// ---------------------------------------------------------------------------
__global__ __launch_bounds__(256) void transpose_cvt(const float* __restrict__ in,
                                                     unsigned short* __restrict__ out_,
                                                     int R, int C) {
    __bf16* out = (__bf16*)out_;
    __shared__ float tile[32][33];
    int bx = blockIdx.x * 32;   // C base
    int by = blockIdx.y * 32;   // R base
    int tx = threadIdx.x, ty = threadIdx.y;  // 32 x 8
#pragma unroll
    for (int i = 0; i < 32; i += 8)
        tile[ty + i][tx] = in[(size_t)(by + ty + i) * C + bx + tx];
    __syncthreads();
#pragma unroll
    for (int i = 0; i < 32; i += 8)
        out[(size_t)(bx + ty + i) * R + by + tx] = (__bf16)tile[tx][ty + i];
}

// ---------------------------------------------------------------------------
// Build combined head weights: Wht[16][1024] bf16 from Wc[1024][4], Wr[1024][12]
// ---------------------------------------------------------------------------
__global__ __launch_bounds__(256) void build_head(const float* __restrict__ Wc,
                                                  const float* __restrict__ Wr,
                                                  unsigned short* __restrict__ out_) {
    __bf16* out = (__bf16*)out_;
    int idx = blockIdx.x * 256 + threadIdx.x;   // 16384
    int n = idx >> 10, k = idx & 1023;
    float v = (n < 4) ? Wc[k * 4 + n] : Wr[k * 12 + (n - 4)];
    out[idx] = (__bf16)v;
}

// ---------------------------------------------------------------------------
// Tiled WMMA GEMM:  C[M][Nc] = relu(A[M][K] @ Bt[Nc][K]^T + bias), bf16 in/out,
// fp32 accumulate. 128x128 tile, 8 waves (2x4), wave tile 64x32, BK=32,
// double-buffered async-to-LDS staging.
// ---------------------------------------------------------------------------
__global__ __launch_bounds__(256) void gemm_bias_relu_bf16(
        const unsigned short* __restrict__ A_,   // [M][K]   bf16
        const unsigned short* __restrict__ Bt_,  // [Nc][K]  bf16 (pre-transposed)
        const float*          __restrict__ bias, // [Nc]
        unsigned short*       __restrict__ C_,   // [M][Nc]  bf16
        int Nc, int K) {
    const __bf16* A  = (const __bf16*)A_;
    const __bf16* Bt = (const __bf16*)Bt_;
    __bf16*       C  = (__bf16*)C_;

    __shared__ __align__(16) __bf16 As[2][128 * 40];
    __shared__ __align__(16) __bf16 Bs[2][128 * 40];

    const int tid  = threadIdx.x;
    const int lane = tid & 31;
    const int wid  = tid >> 5;
    const int wm   = wid >> 2;            // 0..1 -> 64-row half
    const int wn   = wid & 3;             // 0..3 -> 32-col quarter
    const int lr   = lane & 15;           // row (A) / col (B,D) within fragment
    const int kb   = (lane >> 4) << 3;    // K sub-offset 0 or 8
    const int hi8  = (lane >> 4) << 3;    // M offset 0 or 8 for C/D rows

    const __bf16* Ab = A  + (size_t)blockIdx.y * 128 * (size_t)K;
    const __bf16* Bb = Bt + (size_t)blockIdx.x * 128 * (size_t)K;
    const int rowBytes = K * 2;

    const unsigned asB0 = lds_off(&As[0][0]), asB1 = lds_off(&As[1][0]);
    const unsigned bsB0 = lds_off(&Bs[0][0]), bsB1 = lds_off(&Bs[1][0]);

    v8f acc[4][2] = {};

    const int nk = K >> 5;

    // prologue: stage k-step 0 into buffer 0
    issue_tile(Ab, rowBytes, asB0, 0, tid);
    issue_tile(Bb, rowBytes, bsB0, 0, tid);

    for (int kk = 0; kk < nk; ++kk) {
        const int cur = kk & 1;
        if (kk + 1 < nk) {
            const int kbytes = (kk + 1) * 64;
            issue_tile(Ab, rowBytes, cur ? asB0 : asB1, kbytes, tid);
            issue_tile(Bb, rowBytes, cur ? bsB0 : bsB1, kbytes, tid);
            // 4 newer copies in flight; wait until only those remain
            asm volatile("s_wait_asynccnt 0x4" ::: "memory");
        } else {
            asm volatile("s_wait_asynccnt 0x0" ::: "memory");
        }
        __syncthreads();

        const __bf16* Ap = &As[cur][(wm * 64 + lr) * 40 + kb];
        const __bf16* Bp = &Bs[cur][(wn * 32 + lr) * 40 + kb];

        v16bf af[4], bfr[2];
#pragma unroll
        for (int mf = 0; mf < 4; ++mf) af[mf] = load_frag(Ap + mf * 16 * 40);
#pragma unroll
        for (int nf = 0; nf < 2; ++nf) bfr[nf] = load_frag(Bp + nf * 16 * 40);

#pragma unroll
        for (int mf = 0; mf < 4; ++mf)
#pragma unroll
            for (int nf = 0; nf < 2; ++nf)
                acc[mf][nf] = __builtin_amdgcn_wmma_f32_16x16x32_bf16(
                    false, af[mf], false, bfr[nf], (short)0, acc[mf][nf], false, false);

        __syncthreads();   // everyone done reading `cur` before it is re-staged
    }

    // epilogue: bias + relu + bf16 store
    const int rowBase = blockIdx.y * 128 + wm * 64;
    const int colBase = blockIdx.x * 128 + wn * 32;
#pragma unroll
    for (int nf = 0; nf < 2; ++nf) {
        const int col = colBase + nf * 16 + lr;
        const float bv = bias[col];
#pragma unroll
        for (int mf = 0; mf < 4; ++mf) {
#pragma unroll
            for (int r = 0; r < 8; ++r) {
                const int row = rowBase + mf * 16 + hi8 + r;
                float v = acc[mf][nf][r] + bv;
                v = v > 0.f ? v : 0.f;
                C[(size_t)row * Nc + col] = (__bf16)v;
            }
        }
    }
}

// ---------------------------------------------------------------------------
// Head: out = h2[8192,1024] @ Wht[16,1024]^T + bias. One 16x16 WMMA tile/wave,
// K loop of 32. Splits cols 0..3 -> class_logits, 4..15 -> box_pred.
// ---------------------------------------------------------------------------
__global__ __launch_bounds__(256) void head_kernel(
        const unsigned short* __restrict__ h_,    // [N][1024] bf16
        const unsigned short* __restrict__ Wht_,  // [16][1024] bf16
        const float* __restrict__ bc, const float* __restrict__ br,
        float* __restrict__ out, int N) {
    const __bf16* h   = (const __bf16*)h_;
    const __bf16* Wht = (const __bf16*)Wht_;

    const int tid = threadIdx.x, lane = tid & 31, wid = tid >> 5;
    const int lr  = lane & 15;
    const int kb  = (lane >> 4) << 3;
    const int hi8 = (lane >> 4) << 3;
    const int rowBase = (blockIdx.x * 8 + wid) * 16;

    v8f acc = {};
    const __bf16* Ap = h   + (size_t)(rowBase + lr) * 1024 + kb;
    const __bf16* Bp = Wht + (size_t)lr * 1024 + kb;

#pragma unroll 4
    for (int k0 = 0; k0 < 1024; k0 += 32) {
        v16bf a = load_frag(Ap + k0);
        v16bf b = load_frag(Bp + k0);
        acc = __builtin_amdgcn_wmma_f32_16x16x32_bf16(
            false, a, false, b, (short)0, acc, false, false);
    }

    const int n = lr;
    const float bv = (n < 4) ? bc[n] : br[n - 4];
    float* cls = out;
    float* box = out + (size_t)N * 4;
#pragma unroll
    for (int r = 0; r < 8; ++r) {
        int row = rowBase + hi8 + r;
        float v = acc[r] + bv;
        if (n < 4) cls[row * 4 + n] = v;
        else       box[row * 12 + (n - 4)] = v;
    }
}

// ---------------------------------------------------------------------------
// Launch
// ---------------------------------------------------------------------------
extern "C" void kernel_launch(void* const* d_in, const int* in_sizes, int n_in,
                              void* d_out, int out_size, void* d_ws, size_t ws_size,
                              hipStream_t stream) {
    (void)in_sizes; (void)n_in; (void)out_size; (void)ws_size;
    const float* X  = (const float*)d_in[0];
    const float* W1 = (const float*)d_in[1];
    const float* b1 = (const float*)d_in[2];
    const float* W2 = (const float*)d_in[3];
    const float* b2 = (const float*)d_in[4];
    const float* Wc = (const float*)d_in[5];
    const float* bc = (const float*)d_in[6];
    const float* Wr = (const float*)d_in[7];
    const float* br = (const float*)d_in[8];
    float* out = (float*)d_out;

    const int Nn = 8192, D = 12544, H = 1024;

    char* ws = (char*)d_ws;
    size_t off = 0;
    unsigned short* Xb  = (unsigned short*)(ws + off); off += (size_t)Nn * D * 2;
    unsigned short* W1t = (unsigned short*)(ws + off); off += (size_t)D * H * 2;
    unsigned short* W2t = (unsigned short*)(ws + off); off += (size_t)H * H * 2;
    unsigned short* Wht = (unsigned short*)(ws + off); off += (size_t)16 * H * 2;
    unsigned short* h1  = (unsigned short*)(ws + off); off += (size_t)Nn * H * 2;
    unsigned short* h2  = (unsigned short*)(ws + off);

    // 1) precision conversion / weight transposition passes
    long n4 = (long)Nn * D / 4;
    cvt_bf16_4<<<(unsigned)((n4 + 255) / 256), 256, 0, stream>>>(X, Xb, n4);
    transpose_cvt<<<dim3(H / 32, D / 32), dim3(32, 8), 0, stream>>>(W1, W1t, D, H);
    transpose_cvt<<<dim3(H / 32, H / 32), dim3(32, 8), 0, stream>>>(W2, W2t, H, H);
    build_head<<<64, 256, 0, stream>>>(Wc, Wr, Wht);

    // 2) h1 = relu(X @ W1 + b1)   [8192,1024]
    gemm_bias_relu_bf16<<<dim3(H / 128, Nn / 128), 256, 0, stream>>>(Xb, W1t, b1, h1, H, D);
    // 3) h2 = relu(h1 @ W2 + b2)  [8192,1024]
    gemm_bias_relu_bf16<<<dim3(H / 128, Nn / 128), 256, 0, stream>>>(h1, W2t, b2, h2, H, H);
    // 4) heads
    head_kernel<<<Nn / 16 / 8, 256, 0, stream>>>(h2, Wht, bc, br, out, Nn);
}
